// MultiHeadAttention_48911087567389
// MI455X (gfx1250) — compile-verified, matrix operand order checked by reading, not run on hardware
//
#include <hip/hip_runtime.h>

// ---------------------------------------------------------------------------
// MHA forward for MI455X (gfx1250, wave32, WMMA 16x16x32 bf16, f32 accum)
// B=4, S=2048, D=2048, H=16, d_k=128
// ---------------------------------------------------------------------------

typedef __attribute__((ext_vector_type(16))) __bf16 v16bf;
typedef __attribute__((ext_vector_type(8)))  __bf16 v8bf;
typedef __attribute__((ext_vector_type(8)))  float  v8f;
typedef int v4i_vs __attribute__((vector_size(16)));  // matches builtin proto

#if __has_builtin(__builtin_amdgcn_global_load_async_to_lds_b128) && \
    __has_builtin(__builtin_amdgcn_s_wait_asynccnt)
#define ASYNC_LDS 1
#else
#define ASYNC_LDS 0
#endif

__device__ inline __bf16 f2bf(float f) {
  union { float f; unsigned u; } in{f};
  unsigned u = in.u;
  unsigned short r = (unsigned short)((u + 0x7FFFu + ((u >> 16) & 1u)) >> 16);
  union { unsigned short s; __bf16 b; } out{r};
  return out.b;
}

__device__ inline void store_c(float* p, float v)  { *p = v; }
__device__ inline void store_c(__bf16* p, float v) { *p = f2bf(v); }

// stage one 16B chunk global -> LDS (async if available)
__device__ inline void stage16(const __bf16* g, __bf16* l) {
#if ASYNC_LDS
  __builtin_amdgcn_global_load_async_to_lds_b128(
      (__attribute__((address_space(1))) v4i_vs*)g,
      (__attribute__((address_space(3))) v4i_vs*)l, 0, 0);
#else
  *(v8bf*)l = *(const v8bf*)g;
#endif
}

__device__ inline void stage_wait() {
#if ASYNC_LDS
  __builtin_amdgcn_s_wait_asynccnt(0);
#endif
  __syncthreads();
}

// ---------------------------------------------------------------------------
// fp32 -> bf16 conversion (with optional fused scale)
// ---------------------------------------------------------------------------
__global__ void cvt_bf16(const float* __restrict__ in, __bf16* __restrict__ out,
                         size_t n, float scale) {
  size_t i = (size_t)blockIdx.x * 256 + threadIdx.x;
  if (i < n) out[i] = f2bf(in[i] * scale);
}

// ---------------------------------------------------------------------------
// NT GEMM: C[m,n] = sum_k A[m,k] * Bm[n,k] + bias[n]*bias_scale
// A: (M,K) bf16 row-major;  Bm: (N,K) bf16 row-major (nn.Linear weight)
// Block 256 thr = 8 waves; block tile 128x256, BK=32; wave tile 64x64
// (16 WMMAs / k-step). Double-buffered LDS, async global->LDS staging.
// VTRANS: scatter output into per-head-transposed V layout [b][h][dk][s'].
// ---------------------------------------------------------------------------
template <typename OutT, bool VTRANS>
__global__ __launch_bounds__(256)
void gemm_nt(const __bf16* __restrict__ A, const __bf16* __restrict__ Bm,
             const float* __restrict__ bias, float bias_scale,
             OutT* __restrict__ C, int M, int N, int K) {
  __shared__ __align__(16) __bf16 As[2][128][32];   // 16 KB
  __shared__ __align__(16) __bf16 Bs[2][256][32];   // 32 KB

  const int tid   = threadIdx.x;
  const int lane  = tid & 31;
  const int wave  = tid >> 5;
  const int lrow  = lane & 15;
  const int lhalf = lane >> 4;
  const int m0 = blockIdx.y * 128;
  const int n0 = blockIdx.x * 256;
  const int wm = (wave >> 2) * 64;   // 0 / 64
  const int wn = (wave & 3)  * 64;   // 0,64,128,192

  v8f vz = {};
  v8f acc[4][4];
#pragma unroll
  for (int i = 0; i < 4; ++i)
#pragma unroll
    for (int j = 0; j < 4; ++j) acc[i][j] = vz;

  auto stage_tile = [&](int kt, int p) {
    const int k0 = kt * 32;
#pragma unroll
    for (int i = 0; i < 2; ++i) {           // A: 128x32 = 512 chunks
      int id = tid + i * 256;
      int r = id >> 2, c8 = (id & 3) * 8;
      stage16(&A[(size_t)(m0 + r) * K + k0 + c8], &As[p][r][c8]);
    }
#pragma unroll
    for (int i = 0; i < 4; ++i) {           // B: 256x32 = 1024 chunks
      int id = tid + i * 256;
      int r = id >> 2, c8 = (id & 3) * 8;
      stage16(&Bm[(size_t)(n0 + r) * K + k0 + c8], &Bs[p][r][c8]);
    }
  };

  const int nk = K >> 5;
  stage_tile(0, 0);
  stage_wait();

  for (int kt = 0; kt < nk; ++kt) {
    const int p = kt & 1;
    if (kt + 1 < nk) stage_tile(kt + 1, p ^ 1);  // prefetch overlaps WMMAs

    v16bf a_op[4], b_op[4];
#pragma unroll
    for (int i = 0; i < 4; ++i) {
      int r = wm + i * 16 + lrow;
      ((v8bf*)&a_op[i])[0] = *(const v8bf*)&As[p][r][lhalf * 8];
      ((v8bf*)&a_op[i])[1] = *(const v8bf*)&As[p][r][16 + lhalf * 8];
    }
#pragma unroll
    for (int j = 0; j < 4; ++j) {
      int r = wn + j * 16 + lrow;
      ((v8bf*)&b_op[j])[0] = *(const v8bf*)&Bs[p][r][lhalf * 8];
      ((v8bf*)&b_op[j])[1] = *(const v8bf*)&Bs[p][r][16 + lhalf * 8];
    }
#pragma unroll
    for (int i = 0; i < 4; ++i)
#pragma unroll
      for (int j = 0; j < 4; ++j)
        acc[i][j] = __builtin_amdgcn_wmma_f32_16x16x32_bf16(
            false, a_op[i], false, b_op[j], (short)0, acc[i][j], false, false);
    stage_wait();
  }

#pragma unroll
  for (int j = 0; j < 4; ++j) {
    int n = n0 + wn + j * 16 + lrow;
    float bv = bias[n] * bias_scale;
#pragma unroll
    for (int i = 0; i < 4; ++i) {
      int mbase = m0 + wm + i * 16 + lhalf * 8;
#pragma unroll
      for (int r = 0; r < 8; ++r) {
        float v = acc[i][j][r] + bv;
        int m = mbase + r;
        if constexpr (VTRANS) {
          // head-view: h = s>>7, s' = (s&127)*16 + n>>7, c = n&127
          int b = m >> 11, s = m & 2047;
          int h = s >> 7;
          int sp = ((s & 127) << 4) | (n >> 7);
          int c = n & 127;
          C[((size_t)((b << 4) + h) * 128 + c) * 2048 + sp] = f2bf(v);
        } else {
          store_c(&C[(size_t)m * N + n], v);
        }
      }
    }
  }
}

// ---------------------------------------------------------------------------
// Flash attention (bf16 WMMA, online softmax, double-buffered K/V tiles).
// qp/kp: contiguous (B,S,D) == view (B,H,S,128); vt: [b][h][128][S] transposed.
// Scores arrive pre-scaled (scale folded into Wq/bq).
// Output written as (B, S, H*128) bf16 for the final Wo GEMM.
// ---------------------------------------------------------------------------
__global__ __launch_bounds__(256)
void flash_attn(const __bf16* __restrict__ Qh, const __bf16* __restrict__ Kh,
                const __bf16* __restrict__ Vt, __bf16* __restrict__ Xb,
                int Hn, int Sn) {
  __shared__ __align__(16) __bf16 Ks[2][64][128];   // [key][dk]
  __shared__ __align__(16) __bf16 Vs[2][128][64];   // [dk][key]
  __shared__ __align__(16) __bf16 Ps[8][16][64];    // per-wave P scratch

  const int bh = blockIdx.x;
  const int b = bh / Hn, h = bh % Hn;
  const int tid = threadIdx.x, lane = tid & 31, wave = tid >> 5;
  const int lrow = lane & 15, lhalf = lane >> 4;
  const size_t headoff = ((size_t)b * Hn + h) * (size_t)Sn * 128;
  const int q0 = blockIdx.y * 128 + wave * 16;

  // Q tile in A-operand layout (resident for whole kernel)
  v16bf qv[4];
  {
    const __bf16* qrow = Qh + headoff + (size_t)(q0 + lrow) * 128;
#pragma unroll
    for (int c = 0; c < 4; ++c) {
      ((v8bf*)&qv[c])[0] = *(const v8bf*)&qrow[c * 32 + lhalf * 8];
      ((v8bf*)&qv[c])[1] = *(const v8bf*)&qrow[c * 32 + 16 + lhalf * 8];
    }
  }

  v8f vz = {};
  v8f accd[8];
#pragma unroll
  for (int i = 0; i < 8; ++i) accd[i] = vz;
  float m_i[8], l_i[8];
#pragma unroll
  for (int r = 0; r < 8; ++r) { m_i[r] = -1e30f; l_i[r] = 0.f; }

  auto stage_kv = [&](int jt, int p) {
    const int j0 = jt * 64;
#pragma unroll
    for (int i = 0; i < 4; ++i) {
      int id = tid + i * 256;
      int rk = id >> 4, ck = (id & 15) * 8;         // K: 64 rows x 128
      stage16(&Kh[headoff + (size_t)(j0 + rk) * 128 + ck], &Ks[p][rk][ck]);
      int rv = id >> 3, cv = (id & 7) * 8;          // V: 128 rows x 64
      stage16(&Vt[headoff + (size_t)rv * Sn + j0 + cv], &Vs[p][rv][cv]);
    }
  };

  const int nj = Sn >> 6;  // 32
  stage_kv(0, 0);
  stage_wait();

  for (int jt = 0; jt < nj; ++jt) {
    const int p = jt & 1;
    if (jt + 1 < nj) stage_kv(jt + 1, p ^ 1);  // prefetch next K/V tiles

    // ---- S = q . k^T : kk outer, 4 independent WMMAs per step -------------
    v8f sc[4];
#pragma unroll
    for (int kc = 0; kc < 4; ++kc) sc[kc] = vz;
#pragma unroll
    for (int kk = 0; kk < 4; ++kk) {
      v16bf bop[4];
#pragma unroll
      for (int kc = 0; kc < 4; ++kc) {
        const __bf16* krow = &Ks[p][kc * 16 + lrow][kk * 32];
        ((v8bf*)&bop[kc])[0] = *(const v8bf*)&krow[lhalf * 8];
        ((v8bf*)&bop[kc])[1] = *(const v8bf*)&krow[16 + lhalf * 8];
      }
#pragma unroll
      for (int kc = 0; kc < 4; ++kc)
        sc[kc] = __builtin_amdgcn_wmma_f32_16x16x32_bf16(
            false, qv[kk], false, bop[kc], (short)0, sc[kc], false, false);
    }

    // ---- online softmax (row stats per-lane in C-matrix layout) -----------
#pragma unroll
    for (int r = 0; r < 8; ++r) {
      float mx = fmaxf(fmaxf(sc[0][r], sc[1][r]), fmaxf(sc[2][r], sc[3][r]));
#pragma unroll
      for (int d = 1; d < 16; d <<= 1) mx = fmaxf(mx, __shfl_xor(mx, d, 32));
      float newm = fmaxf(m_i[r], mx);
      float corr = __expf(m_i[r] - newm);
      m_i[r] = newm;
      float rs = 0.f;
#pragma unroll
      for (int kc = 0; kc < 4; ++kc) {
        float pj = __expf(sc[kc][r] - newm);
        sc[kc][r] = pj;
        rs += pj;
      }
#pragma unroll
      for (int d = 1; d < 16; d <<= 1) rs += __shfl_xor(rs, d, 32);
      l_i[r] = l_i[r] * corr + rs;
#pragma unroll
      for (int i = 0; i < 8; ++i) accd[i][r] *= corr;
    }

    // ---- P: C-layout regs -> A-layout via per-wave LDS --------------------
#pragma unroll
    for (int kc = 0; kc < 4; ++kc)
#pragma unroll
      for (int r = 0; r < 8; ++r)
        Ps[wave][r + lhalf * 8][kc * 16 + lrow] = f2bf(sc[kc][r]);

    // ---- O += P . V : groups of 4 independent WMMAs -----------------------
#pragma unroll
    for (int ks = 0; ks < 2; ++ks) {
      v16bf pop;
      const __bf16* prow = &Ps[wave][lrow][ks * 32];
      ((v8bf*)&pop)[0] = *(const v8bf*)&prow[lhalf * 8];
      ((v8bf*)&pop)[1] = *(const v8bf*)&prow[16 + lhalf * 8];
#pragma unroll
      for (int dg = 0; dg < 2; ++dg) {
        v16bf vop[4];
#pragma unroll
        for (int t = 0; t < 4; ++t) {
          const __bf16* vrow = &Vs[p][(dg * 4 + t) * 16 + lrow][ks * 32];
          ((v8bf*)&vop[t])[0] = *(const v8bf*)&vrow[lhalf * 8];
          ((v8bf*)&vop[t])[1] = *(const v8bf*)&vrow[16 + lhalf * 8];
        }
#pragma unroll
        for (int t = 0; t < 4; ++t)
          accd[dg * 4 + t] = __builtin_amdgcn_wmma_f32_16x16x32_bf16(
              false, pop, false, vop[t], (short)0, accd[dg * 4 + t], false,
              false);
      }
    }
    stage_wait();
  }

  // epilogue: normalize and scatter into (B, S, H*128)
  const int Dm = Hn * 128;
#pragma unroll
  for (int dc = 0; dc < 8; ++dc) {
    int col = h * 128 + dc * 16 + lrow;
#pragma unroll
    for (int r = 0; r < 8; ++r) {
      int qr = q0 + lhalf * 8 + r;
      Xb[((size_t)b * Sn + qr) * Dm + col] = f2bf(accd[dc][r] / l_i[r]);
    }
  }
}

// ---------------------------------------------------------------------------
// Host launcher
// ---------------------------------------------------------------------------
extern "C" void kernel_launch(void* const* d_in, const int* in_sizes, int n_in,
                              void* d_out, int out_size, void* d_ws,
                              size_t ws_size, hipStream_t stream) {
  const int Bn = 4, Sn = 2048, Dm = 2048, Hn = 16;
  const size_t SD = (size_t)Bn * Sn * Dm;  // 16,777,216
  const size_t DD = (size_t)Dm * Dm;       //  4,194,304
  const float qscale = 0.08838834764831845f;  // 1/sqrt(128)

  const float* Q  = (const float*)d_in[0];
  const float* K  = (const float*)d_in[1];
  const float* V  = (const float*)d_in[2];
  const float* Wq = (const float*)d_in[3];
  const float* bq = (const float*)d_in[4];
  const float* Wk = (const float*)d_in[5];
  const float* bk = (const float*)d_in[6];
  const float* Wv = (const float*)d_in[7];
  const float* bv = (const float*)d_in[8];
  const float* Wo = (const float*)d_in[9];
  const float* bo = (const float*)d_in[10];

  __bf16* w   = (__bf16*)d_ws;
  __bf16* Qb  = w;            __bf16* Kb  = Qb + SD;  __bf16* Vb  = Kb + SD;
  __bf16* Wqb = Vb + SD;      __bf16* Wkb = Wqb + DD;
  __bf16* Wvb = Wkb + DD;     __bf16* Wob = Wvb + DD;
  __bf16* qp  = Wob + DD;     __bf16* kp  = qp + SD;
  __bf16* vt  = kp + SD;      __bf16* xb  = vt + SD;

  const int blkSD = (int)((SD + 255) / 256);
  const int blkDD = (int)((DD + 255) / 256);
  cvt_bf16<<<blkSD, 256, 0, stream>>>(Q, Qb, SD, 1.0f);
  cvt_bf16<<<blkSD, 256, 0, stream>>>(K, Kb, SD, 1.0f);
  cvt_bf16<<<blkSD, 256, 0, stream>>>(V, Vb, SD, 1.0f);
  cvt_bf16<<<blkDD, 256, 0, stream>>>(Wq, Wqb, DD, qscale);  // fold 1/sqrt(dk)
  cvt_bf16<<<blkDD, 256, 0, stream>>>(Wk, Wkb, DD, 1.0f);
  cvt_bf16<<<blkDD, 256, 0, stream>>>(Wv, Wvb, DD, 1.0f);
  cvt_bf16<<<blkDD, 256, 0, stream>>>(Wo, Wob, DD, 1.0f);

  const int M = Bn * Sn;  // 8192
  dim3 gg(Dm / 256, M / 128);  // (8, 64)
  gemm_nt<__bf16, false><<<gg, 256, 0, stream>>>(Qb, Wqb, bq, qscale, qp, M, Dm, Dm);
  gemm_nt<__bf16, false><<<gg, 256, 0, stream>>>(Kb, Wkb, bk, 1.0f, kp, M, Dm, Dm);
  gemm_nt<__bf16, true ><<<gg, 256, 0, stream>>>(Vb, Wvb, bv, 1.0f, vt, M, Dm, Dm);

  dim3 ga(Bn * Hn, Sn / 128);  // (64, 16)
  flash_attn<<<ga, 256, 0, stream>>>(qp, kp, vt, xb, Hn, Sn);

  gemm_nt<float, false><<<gg, 256, 0, stream>>>(xb, Wob, bo, 1.0f,
                                                (float*)d_out, M, Dm, Dm);
}
